// QwenModel_24867860644065
// MI455X (gfx1250) — compile-verified, matrix operand order checked by reading, not run on hardware
//
#include <hip/hip_runtime.h>
#include <hip/hip_bf16.h>

// ---------------- problem constants ----------------
#define BB     2
#define SS     1024
#define HH     2560
#define NH     32
#define NKV    8
#define HD     128
#define INTER  9728
#define GROUP  128
#define EPS    1e-6f
#define THETA  1000000.0f
#define TT     (BB * SS)          // 2048 tokens
#define QD     (NH * HD)          // 4096
#define KVD    (NKV * HD)         // 1024

typedef __attribute__((ext_vector_type(16))) _Float16 v16h;
typedef __attribute__((ext_vector_type(8)))  float    v8f;
typedef __attribute__((ext_vector_type(4)))  int      v4i;

// gfx1250 async global->LDS copy (ASYNCcnt-tracked), if this toolchain has it.
// Probe result (round 2 diagnostic): param 0 is a generic pointer to a
// 4 x i32 vector, so pass plain v4i* for both global src and LDS dst.
#if __has_builtin(__builtin_amdgcn_global_load_async_to_lds_b128)
#define HAVE_ASYNC_LDS 1
__device__ __forceinline__ void async_copy16(const void* g, void* l) {
    __builtin_amdgcn_global_load_async_to_lds_b128(
        (v4i*)g, (v4i*)l, /*offset=*/0, /*cpol=*/0);
}
__device__ __forceinline__ void wait_async0() {
#if __has_builtin(__builtin_amdgcn_s_wait_asynccnt)
    __builtin_amdgcn_s_wait_asynccnt(0);
#else
    asm volatile("s_wait_asynccnt 0" ::: "memory");
#endif
}
#endif

// K index mapping for 16-bit A/B WMMA fragments (ISA 7.12.2):
// VGPR j holds halves (2j, 2j+1); j<4 -> K base 2j, j>=4 -> K base 16+2(j-4);
// half-wave (lane>=16) offsets K by +8.
__device__ __forceinline__ int kmap(int j, int kh) {
    return ((j < 4) ? (2 * j) : (16 + 2 * (j - 4))) + 8 * kh;
}

__device__ __forceinline__ v8f wmma_f16(v16h a, v16h b, v8f c) {
    return __builtin_amdgcn_wmma_f32_16x16x32_f16(
        /*neg_a=*/false, a, /*neg_b=*/false, b,
        /*c_mod=*/(short)0, c, /*reuse_a=*/false, /*reuse_b=*/false);
}

// ---------------- AWQ int4 dequant + WMMA GEMM ----------------
// C(MxN) = A(MxK, f16 row-major) x dequant(qw, sc, qz)(KxN)
// Block: 256 threads (8 waves). Tile: BM=64, BN=64, K-step 32.
// A tile is moved with GLOBAL_LOAD_ASYNC_TO_LDS_B128 when available
// (16B per lane, ASYNCcnt), W tile is nibble-dequanted into LDS by VALU.
template <bool F16OUT>
__global__ __launch_bounds__(256)
void awq_gemm_kernel(const _Float16* __restrict__ A,
                     const int*      __restrict__ qw,   // (K/8) x N
                     const float*    __restrict__ sc,   // (K/GROUP) x N
                     const int*      __restrict__ qz,   // (K/GROUP) x N
                     float*          __restrict__ Cf,
                     _Float16*       __restrict__ Ch,
                     int M, int N, int K) {
    __shared__ _Float16 As[64][40];   // 64 rows x 32 K (padded; row stride 80B)
    __shared__ _Float16 Ws[32][72];   // 32 K rows x 64 N cols (padded)

    const int tid   = threadIdx.x;
    const int lane  = tid & 31;
    const int wave  = tid >> 5;
    const int wm    = wave & 3;
    const int wn    = wave >> 2;
    const int l15   = lane & 15;
    const int khalf = lane >> 4;

    const int mb = blockIdx.y * 64;
    const int nb = blockIdx.x * 64;

    // A cooperative load: thread -> one 16B chunk (8 halves)
    const int arow  = tid >> 2;          // 0..63
    const int acol8 = (tid & 3) * 8;     // 0,8,16,24
    // W dequant: thread -> one packed int32 (8 K-rows) of one column
    const int qr = tid >> 6;             // 0..3
    const int wc = tid & 63;             // 0..63

    v8f acc0 = {}, acc1 = {};

    const int nsteps = K / 32;
    for (int s = 0; s < nsteps; ++s) {
        const int kb = s * 32;
        // ---- A tile -> LDS ----
        {
            const _Float16* src = A + (size_t)(mb + arow) * K + kb + acol8;
#if defined(HAVE_ASYNC_LDS)
            async_copy16(src, &As[arow][acol8]);
#else
            uint4 u = *(const uint4*)src;
            const _Float16* hp = (const _Float16*)&u;
#pragma unroll
            for (int i = 0; i < 8; ++i) As[arow][acol8 + i] = hp[i];
#endif
            if (s + 1 < nsteps)
                __builtin_prefetch(src + 32, 0, 3);   // global_prefetch_b8
        }
        // ---- dequant W tile -> LDS ----
        {
            const int   g   = kb / GROUP;       // step fully inside one group
            const int   col = nb + wc;
            const float zf  = (float)qz[(size_t)g * N + col];
            const float sv  = sc[(size_t)g * N + col];
            const int   pk  = qw[(size_t)(kb / 8 + qr) * N + col];
#pragma unroll
            for (int nib = 0; nib < 8; ++nib) {
                float w = (float)((pk >> (4 * nib)) & 15);
                Ws[qr * 8 + nib][wc] = (_Float16)((w - zf) * sv);
            }
        }
#if defined(HAVE_ASYNC_LDS)
        wait_async0();                 // this wave's async LDS writes landed
#endif
        __syncthreads();               // all waves' tiles visible

        // ---- fragments ----
        v16h a;
#pragma unroll
        for (int j = 0; j < 8; ++j) {
            const int off = kmap(j, khalf);
            a[2 * j]     = As[wm * 16 + l15][off];
            a[2 * j + 1] = As[wm * 16 + l15][off + 1];
        }
        v16h b0, b1;
#pragma unroll
        for (int j = 0; j < 8; ++j) {
            const int off = kmap(j, khalf);
            b0[2 * j]     = Ws[off][wn * 32 + l15];
            b0[2 * j + 1] = Ws[off + 1][wn * 32 + l15];
            b1[2 * j]     = Ws[off][wn * 32 + 16 + l15];
            b1[2 * j + 1] = Ws[off + 1][wn * 32 + 16 + l15];
        }
        acc0 = wmma_f16(a, b0, acc0);
        acc1 = wmma_f16(a, b1, acc1);
        __syncthreads();
    }

    // ---- store C (ISA C layout: N per lane, M in VGPRs, +8 rows for hi half) ----
    const int col0    = nb + wn * 32 + l15;
    const int col1    = col0 + 16;
    const int rowbase = mb + wm * 16 + 8 * khalf;
#pragma unroll
    for (int r = 0; r < 8; ++r) {
        const size_t row = (size_t)(rowbase + r);
        if constexpr (F16OUT) {
            Ch[row * N + col0] = (_Float16)acc0[r];
            Ch[row * N + col1] = (_Float16)acc1[r];
        } else {
            Cf[row * N + col0] = acc0[r];
            Cf[row * N + col1] = acc1[r];
        }
    }
}

// ---------------- RMSNorm: f32 in -> f16 out ----------------
__global__ __launch_bounds__(256)
void rmsnorm_kernel(const float* __restrict__ x, const float* __restrict__ w,
                    _Float16* __restrict__ out) {
    __shared__ float red[256];
    const int row = blockIdx.x;
    const float* xr = x + (size_t)row * HH;
    float ss = 0.f;
    for (int i = threadIdx.x; i < HH; i += 256) { float v = xr[i]; ss += v * v; }
    red[threadIdx.x] = ss;
    __syncthreads();
    for (int s = 128; s > 0; s >>= 1) {
        if (threadIdx.x < s) red[threadIdx.x] += red[threadIdx.x + s];
        __syncthreads();
    }
    const float inv = rsqrtf(red[0] / (float)HH + EPS);
    for (int i = threadIdx.x; i < HH; i += 256)
        out[(size_t)row * HH + i] = (_Float16)(w[i] * xr[i] * inv);
}

// ---------------- bias + RoPE, reshape to [b][h][s][d] f16 ----------------
__global__ __launch_bounds__(256)
void rope_kernel(const float* __restrict__ X, const float* __restrict__ bias,
                 _Float16* __restrict__ out, int nheads) {
    const int idx  = blockIdx.x * 256 + threadIdx.x;
    const int cols = nheads * HD;
    const int ntot = TT * cols;
    if (idx >= ntot) return;
    const int d    = idx % HD;
    const int rest = idx / HD;
    const int hh   = rest % nheads;
    const int tok  = rest / nheads;
    const int s    = tok % SS;
    const int b    = tok / SS;
    const int col  = hh * HD + d;

    float val = X[(size_t)tok * cols + col] + bias[col];
    const int   i   = d & 63;
    const float inv = __powf(THETA, -((float)(2 * i)) / (float)HD);
    const float ang = (float)s * inv;
    float sn, cs;
    __sincosf(ang, &sn, &cs);
    float other;
    if (d < 64) {
        other = X[(size_t)tok * cols + col + 64] + bias[col + 64];
        val = val * cs - other * sn;
    } else {
        other = X[(size_t)tok * cols + col - 64] + bias[col - 64];
        val = val * cs + other * sn;
    }
    out[(((size_t)b * nheads + hh) * SS + s) * HD + d] = (_Float16)val;
}

// ---------------- bias + cast V to [b][hk][s][d] f16 ----------------
__global__ __launch_bounds__(256)
void biasv_kernel(const float* __restrict__ X, const float* __restrict__ bias,
                  _Float16* __restrict__ out) {
    const int idx = blockIdx.x * 256 + threadIdx.x;
    const int ntot = TT * KVD;
    if (idx >= ntot) return;
    const int d    = idx % HD;
    const int rest = idx / HD;
    const int hh   = rest % NKV;
    const int tok  = rest / NKV;
    const int s    = tok % SS;
    const int b    = tok / SS;
    const int col  = hh * HD + d;
    out[(((size_t)b * NKV + hh) * SS + s) * HD + d] =
        (_Float16)(X[(size_t)tok * KVD + col] + bias[col]);
}

// ---------------- flash attention, one wave per (b, h, 16-q tile) ----------------
__global__ __launch_bounds__(32)
void attn_kernel(const _Float16* __restrict__ Q, const _Float16* __restrict__ Kt,
                 const _Float16* __restrict__ V, _Float16* __restrict__ O) {
    __shared__ _Float16 Ps[16][40];     // 16 q-rows x 32 key probs (padded)

    const int lane = threadIdx.x;
    const int l15  = lane & 15;
    const int hi   = lane >> 4;
    const int qt   = blockIdx.x;
    const int bh   = blockIdx.y;
    const int b    = bh / NH;
    const int h    = bh % NH;
    const int hk   = h / (NH / NKV);
    const int qbase = qt * 16;
    const float scale = 0.08838834764831845f;   // 1/sqrt(128)

    // Q tile 16x128 as 4 A-fragments (K-steps of 32)
    const _Float16* qp = Q + ((size_t)(b * NH + h) * SS + qbase + l15) * HD;
    v16h aq[4];
#pragma unroll
    for (int s4 = 0; s4 < 4; ++s4) {
#pragma unroll
        for (int j = 0; j < 8; ++j) {
            const int off = kmap(j, hi);
            aq[s4][2 * j]     = qp[s4 * 32 + off];
            aq[s4][2 * j + 1] = qp[s4 * 32 + off + 1];
        }
    }

    float m[8], se[8];
    v8f acc[8] = {};
#pragma unroll
    for (int r = 0; r < 8; ++r) { m[r] = -3.0e38f; se[r] = 0.f; }

    const _Float16* kbp = Kt + (size_t)(b * NKV + hk) * SS * HD;
    const _Float16* vbp = V  + (size_t)(b * NKV + hk) * SS * HD;

    const int nkb = (qbase + 15) / 32 + 1;   // causal: keys <= last q row
    for (int kb = 0; kb < nkb; ++kb) {
        const int k0 = kb * 32;

        // ---- scores S = Q x K^T : two 16x16 tiles over a 32-key block ----
        v8f c0 = {}, c1 = {};
#pragma unroll
        for (int s4 = 0; s4 < 4; ++s4) {
            v16h bk0, bk1;
            const _Float16* kp0 = kbp + (size_t)(k0 + l15) * HD + s4 * 32;
            const _Float16* kp1 = kbp + (size_t)(k0 + 16 + l15) * HD + s4 * 32;
#pragma unroll
            for (int j = 0; j < 8; ++j) {
                const int off = kmap(j, hi);
                bk0[2 * j] = kp0[off]; bk0[2 * j + 1] = kp0[off + 1];
                bk1[2 * j] = kp1[off]; bk1[2 * j + 1] = kp1[off + 1];
            }
            c0 = wmma_f16(aq[s4], bk0, c0);
            c1 = wmma_f16(aq[s4], bk1, c1);
        }

        // ---- online softmax (row stats live in the 16-lane group owning the row) ----
#pragma unroll
        for (int r = 0; r < 8; ++r) {
            const int row  = qbase + r + 8 * hi;
            const int col0 = k0 + l15;
            const int col1 = col0 + 16;
            float s0 = c0[r] * scale; if (col0 > row) s0 = -3.0e38f;
            float s1 = c1[r] * scale; if (col1 > row) s1 = -3.0e38f;
            float mx = fmaxf(s0, s1);
#pragma unroll
            for (int d = 1; d < 16; d <<= 1) mx = fmaxf(mx, __shfl_xor(mx, d, 32));
            const float mnew = fmaxf(m[r], mx);
            const float corr = __expf(m[r] - mnew);
            const float p0 = __expf(s0 - mnew);
            const float p1 = __expf(s1 - mnew);
            float ps = p0 + p1;
#pragma unroll
            for (int d = 1; d < 16; d <<= 1) ps += __shfl_xor(ps, d, 32);
            se[r] = se[r] * corr + ps;
            m[r]  = mnew;
#pragma unroll
            for (int t = 0; t < 8; ++t) acc[t][r] *= corr;
            Ps[r + 8 * hi][l15]      = (_Float16)p0;
            Ps[r + 8 * hi][16 + l15] = (_Float16)p1;
        }
        __syncthreads();

        // ---- re-fragment P (C layout -> A layout) via LDS ----
        v16h pa;
#pragma unroll
        for (int j = 0; j < 8; ++j) {
            const int off = kmap(j, hi);
            pa[2 * j]     = Ps[l15][off];
            pa[2 * j + 1] = Ps[l15][off + 1];
        }
        __syncthreads();

        // ---- O += P x V : 8 head-dim tiles of 16 ----
#pragma unroll
        for (int t = 0; t < 8; ++t) {
            v16h bv;
#pragma unroll
            for (int j = 0; j < 8; ++j) {
                const int off = kmap(j, hi);
                bv[2 * j]     = vbp[(size_t)(k0 + off) * HD + t * 16 + l15];
                bv[2 * j + 1] = vbp[(size_t)(k0 + off + 1) * HD + t * 16 + l15];
            }
            acc[t] = wmma_f16(pa, bv, acc[t]);
        }
    }

    // ---- normalize + write, token-major [tok][h*HD + d] f16 ----
#pragma unroll
    for (int r = 0; r < 8; ++r) {
        const int   row = qbase + r + 8 * hi;
        const float inv = 1.0f / se[r];
        const size_t obase = ((size_t)b * SS + row) * QD + (size_t)h * HD;
#pragma unroll
        for (int t = 0; t < 8; ++t)
            O[obase + t * 16 + l15] = (_Float16)(acc[t][r] * inv);
    }
}

// ---------------- elementwise glue ----------------
__global__ __launch_bounds__(256)
void add_kernel(const float* __restrict__ a, const float* __restrict__ b,
                float* __restrict__ out, int n) {
    const int i = blockIdx.x * 256 + threadIdx.x;
    if (i < n) out[i] = a[i] + b[i];
}

__global__ __launch_bounds__(256)
void swiglu_kernel(const _Float16* __restrict__ g, const _Float16* __restrict__ u,
                   _Float16* __restrict__ out, int n) {
    const int i = blockIdx.x * 256 + threadIdx.x;
    if (i < n) {
        const float gv = (float)g[i];
        const float uv = (float)u[i];
        const float sil = gv * (1.0f / (1.0f + __expf(-gv)));
        out[i] = (_Float16)(sil * uv);
    }
}

// ---------------- launcher ----------------
extern "C" void kernel_launch(void* const* d_in, const int* in_sizes, int n_in,
                              void* d_out, int out_size, void* d_ws, size_t ws_size,
                              hipStream_t stream) {
    (void)in_sizes; (void)n_in; (void)out_size; (void)ws_size;

    const float* x    = (const float*)d_in[0];
    const float* sc_q = (const float*)d_in[1];
    const float* sc_k = (const float*)d_in[2];
    const float* sc_v = (const float*)d_in[3];
    const float* sc_o = (const float*)d_in[4];
    const float* sc_g = (const float*)d_in[5];
    const float* sc_u = (const float*)d_in[6];
    const float* sc_d = (const float*)d_in[7];
    const float* bq   = (const float*)d_in[8];
    const float* bk   = (const float*)d_in[9];
    const float* bv   = (const float*)d_in[10];
    const float* wn1  = (const float*)d_in[11];
    const float* wn2  = (const float*)d_in[12];
    const int* qw_q = (const int*)d_in[13];
    const int* qw_k = (const int*)d_in[14];
    const int* qw_v = (const int*)d_in[15];
    const int* qw_o = (const int*)d_in[16];
    const int* qw_g = (const int*)d_in[17];
    const int* qw_u = (const int*)d_in[18];
    const int* qw_d = (const int*)d_in[19];
    const int* qz_q = (const int*)d_in[20];
    const int* qz_k = (const int*)d_in[21];
    const int* qz_v = (const int*)d_in[22];
    const int* qz_o = (const int*)d_in[23];
    const int* qz_g = (const int*)d_in[24];
    const int* qz_u = (const int*)d_in[25];
    const int* qz_d = (const int*)d_in[26];

    // workspace arena (reused regions; ~208 MB total)
    char* ws = (char*)d_ws;
    size_t o = 0;
    auto take = [&](size_t bytes) {
        size_t r = o;
        o += (bytes + 255) & ~(size_t)255;
        return r;
    };
    _Float16* h16  = (_Float16*)(ws + take((size_t)TT * HH * 2));     // rmsnorm1 then rmsnorm2
    float*    qf32 = (float*)   (ws + take((size_t)TT * QD * 4));     // later reused as attn f16
    float*    kf32 = (float*)   (ws + take((size_t)TT * KVD * 4));
    float*    vf32 = (float*)   (ws + take((size_t)TT * KVD * 4));
    _Float16* Qh   = (_Float16*)(ws + take((size_t)TT * QD * 2));
    _Float16* Kh   = (_Float16*)(ws + take((size_t)TT * KVD * 2));
    _Float16* Vh   = (_Float16*)(ws + take((size_t)TT * KVD * 2));
    float*    of32 = (float*)   (ws + take((size_t)TT * HH * 4));     // later reused as down-proj out
    float*    x1   = (float*)   (ws + take((size_t)TT * HH * 4));
    _Float16* g16  = (_Float16*)(ws + take((size_t)TT * INTER * 2));  // becomes gu in place
    _Float16* u16  = (_Float16*)(ws + take((size_t)TT * INTER * 2));
    _Float16* attnh = (_Float16*)qf32;   // q f32 dead after RoPE
    float*    df32  = of32;              // o f32 dead after residual add

    float* out = (float*)d_out;

    const dim3 blk256(256);

    // 1) RMSNorm 1
    rmsnorm_kernel<<<TT, blk256, 0, stream>>>(x, wn1, h16);

    // 2) QKV projections (int4 dequant + WMMA)
    awq_gemm_kernel<false><<<dim3(QD / 64,  TT / 64), blk256, 0, stream>>>(
        h16, qw_q, sc_q, qz_q, qf32, nullptr, TT, QD, HH);
    awq_gemm_kernel<false><<<dim3(KVD / 64, TT / 64), blk256, 0, stream>>>(
        h16, qw_k, sc_k, qz_k, kf32, nullptr, TT, KVD, HH);
    awq_gemm_kernel<false><<<dim3(KVD / 64, TT / 64), blk256, 0, stream>>>(
        h16, qw_v, sc_v, qz_v, vf32, nullptr, TT, KVD, HH);

    // 3) bias + RoPE + per-head f16 layouts
    rope_kernel<<<(TT * QD + 255) / 256, blk256, 0, stream>>>(qf32, bq, Qh, NH);
    rope_kernel<<<(TT * KVD + 255) / 256, blk256, 0, stream>>>(kf32, bk, Kh, NKV);
    biasv_kernel<<<(TT * KVD + 255) / 256, blk256, 0, stream>>>(vf32, bv, Vh);

    // 4) flash attention (WMMA QK^T and PV)
    attn_kernel<<<dim3(SS / 16, BB * NH), dim3(32), 0, stream>>>(Qh, Kh, Vh, attnh);

    // 5) output projection + residual
    awq_gemm_kernel<false><<<dim3(HH / 64, TT / 64), blk256, 0, stream>>>(
        attnh, qw_o, sc_o, qz_o, of32, nullptr, TT, HH, QD);
    add_kernel<<<(TT * HH + 255) / 256, blk256, 0, stream>>>(x, of32, x1, TT * HH);

    // 6) RMSNorm 2
    rmsnorm_kernel<<<TT, blk256, 0, stream>>>(x1, wn2, h16);

    // 7) SwiGLU MLP
    awq_gemm_kernel<true><<<dim3(INTER / 64, TT / 64), blk256, 0, stream>>>(
        h16, qw_g, sc_g, qz_g, nullptr, g16, TT, INTER, HH);
    awq_gemm_kernel<true><<<dim3(INTER / 64, TT / 64), blk256, 0, stream>>>(
        h16, qw_u, sc_u, qz_u, nullptr, u16, TT, INTER, HH);
    swiglu_kernel<<<(TT * INTER + 255) / 256, blk256, 0, stream>>>(g16, u16, g16, TT * INTER);
    awq_gemm_kernel<false><<<dim3(HH / 64, TT / 64), blk256, 0, stream>>>(
        g16, qw_d, sc_d, qz_d, df32, nullptr, TT, HH, INTER);

    // 8) final residual -> d_out
    add_kernel<<<(TT * HH + 255) / 256, blk256, 0, stream>>>(x1, df32, out, TT * HH);
}
